// HyperMP_Block_6356551598645
// MI455X (gfx1250) — matrix operand order repaired and checked by reading, not compiled, not run on hardware
//
#include <hip/hip_runtime.h>
#include <hip/hip_bf16.h>

// ---------- types ----------
typedef __attribute__((ext_vector_type(16))) __bf16 v16bf;
typedef __attribute__((ext_vector_type(8)))  float  v8f;
typedef __attribute__((ext_vector_type(4)))  int    v4i;

#define DIMV 256

#if defined(__gfx1250__) && __has_builtin(__builtin_amdgcn_global_load_async_to_lds_b128)
#define HAS_ASYNC 1
typedef __attribute__((address_space(1))) v4i as1_v4i;
typedef __attribute__((address_space(3))) v4i as3_v4i;
#else
#define HAS_ASYNC 0
#endif

__device__ __forceinline__ void async_wait0() {
#if HAS_ASYNC
#if __has_builtin(__builtin_amdgcn_s_wait_asynccnt)
    __builtin_amdgcn_s_wait_asynccnt(0);
#else
    asm volatile("s_wait_asynccnt 0x0" ::: "memory");
#endif
#endif
}

union Frag16 { v16bf v; uint4 q[2]; };
union PkBf4  { __bf16 h[4]; unsigned long long u; };

// ---------------------------------------------------------------------------
// bf16 WMMA GEMM:  out = epilogue( A @ W + bias )
//   ABF=1: A is bf16 (tile staged via async-to-LDS byte copy)
//   ABF=0: A is f32 (register prefetch + hardware cvt to bf16)
//   A optionally split in K at K0 (concat([A0,A1],-1) @ W); tiles never
//   straddle the split. WT: bf16 weights pre-transposed to N x K.
//   OBF=1: out is bf16; OBF=0: out is f32. res (EPI 1/2) is always bf16.
//   EPI 0: out = relu(acc + b)
//   EPI 1: out = relu(res + acc + b)
//   EPI 2: out = res + relu(acc + b)
// Block: 256 threads (8 wave32), tile 128x128, BK=32, double-buffered LDS.
// ---------------------------------------------------------------------------
template <int EPI, int ABF, int OBF>
__global__ void __launch_bounds__(256)
gemm_bf16_wmma(int M, int N, int K0, int K,
               const void* __restrict__ A0v, const void* __restrict__ A1v,
               const unsigned short* __restrict__ WT,
               const float* __restrict__ bias,
               const unsigned short* __restrict__ res,
               void* __restrict__ outv)
{
    __shared__ unsigned short Asl[2][128 * 40];   // [row][k], stride 40 (80B)
    __shared__ unsigned short Bsl[2][128 * 40];   // [n][k]

    const int tid   = threadIdx.x;
    const int wid   = tid >> 5;
    const int lane  = tid & 31;
    const int half  = lane >> 4;
    const int l16   = lane & 15;
    const int waveM = wid & 3;                 // 4 waves along M (32 rows each)
    const int waveN = wid >> 2;                // 2 waves along N (64 cols each)
    const int m0    = blockIdx.x * 128;
    const int n0    = blockIdx.y * 128;
    const int nk    = K >> 5;
    const int rowLimA = M - m0 - 1;            // clamp for M tail

    const int cg4 = tid & 3, r4 = tid >> 2;    // 4x 16B chunks/row, 64 rows/pass
    const int cg8 = tid & 7, r8 = tid >> 3;    // f32 A: 8x float4/row, 32 rows/pass

    // copy one 128x32 bf16 tile (row-major, ld elems) into LDS (async if avail)
    auto copyTile = [&](const unsigned short* src, int ld, unsigned short* dst,
                        int rowLim) {
        #pragma unroll
        for (int p = 0; p < 2; ++p) {
            int row = r4 + p * 64;
            int rr  = row <= rowLim ? row : rowLim;
            const unsigned short* gp = src + (size_t)rr * ld + cg4 * 8;
#if HAS_ASYNC
            __builtin_amdgcn_global_load_async_to_lds_b128(
                (as1_v4i*)(unsigned short*)gp,
                (as3_v4i*)&dst[row * 40 + cg4 * 8], 0, 0);
#else
            *(uint4*)&dst[row * 40 + cg4 * 8] = *(const uint4*)gp;
#endif
        }
    };
    // A-half select (returns element offset within the chosen matrix)
    auto selOff = [&](int kb, int& kk) -> int {   // returns lda
        if (kb < K0) { kk = kb;      return K0; }
        kk = kb - K0;                return K - K0;
    };
    auto stageA_bf = [&](int kb, unsigned short* dst) {
        int kk; int lda = selOff(kb, kk);
        const unsigned short* base =
            (const unsigned short*)(kb < K0 ? A0v : A1v) + (size_t)m0 * lda + kk;
        copyTile(base, lda, dst, rowLimA);
    };
    auto stageB = [&](int kb, unsigned short* dst) {
        copyTile(WT + (size_t)n0 * K + kb, K, dst, 127);
    };

    float4 areg[4];                            // f32 A register prefetch
    auto loadAf = [&](int kb) {
        int kk; int lda = selOff(kb, kk);
        const float* Ap = (const float*)(kb < K0 ? A0v : A1v);
        #pragma unroll
        for (int p = 0; p < 4; ++p) {
            int grow = m0 + r8 + p * 32; if (grow >= M) grow = M - 1;
            areg[p] = *(const float4*)(Ap + (size_t)grow * lda + kk + cg8 * 4);
        }
    };
    auto storeAf = [&](unsigned short* As) {   // hardware cvt f32->bf16
        #pragma unroll
        for (int p = 0; p < 4; ++p) {
            PkBf4 pk;
            pk.h[0] = (__bf16)areg[p].x; pk.h[1] = (__bf16)areg[p].y;
            pk.h[2] = (__bf16)areg[p].z; pk.h[3] = (__bf16)areg[p].w;
            *(unsigned long long*)(&As[(r8 + p * 32) * 40 + cg8 * 4]) = pk.u;
        }
    };

    v8f acc[2][4] = {};

    // ---- prologue: stage tile 0 into buffer 0 ----
    if constexpr (ABF) { stageA_bf(0, Asl[0]); }
    else               { loadAf(0); storeAf(Asl[0]); }
    stageB(0, Bsl[0]);
    async_wait0();
    __syncthreads();

    for (int kt = 0; kt < nk; ++kt) {
        const int cur = kt & 1, nxt = cur ^ 1;
        const bool more = (kt + 1 < nk);

        if (more) {                            // fill next buffer during WMMA
            if constexpr (ABF) stageA_bf((kt + 1) << 5, Asl[nxt]);
            else               loadAf((kt + 1) << 5);
            stageB((kt + 1) << 5, Bsl[nxt]);
        }

        // ---- fragments per ISA layout tables ----
        Frag16 afr[2], bfr[4];
        #pragma unroll
        for (int tm = 0; tm < 2; ++tm) {
            const unsigned short* ap = &Asl[cur][(waveM * 32 + tm * 16 + l16) * 40 + half * 8];
            afr[tm].q[0] = *(const uint4*)ap;          // K = half*8 .. +7
            afr[tm].q[1] = *(const uint4*)(ap + 16);   // K = 16+half*8 .. +7
        }
        #pragma unroll
        for (int tn = 0; tn < 4; ++tn) {
            const unsigned short* bp = &Bsl[cur][(waveN * 64 + tn * 16 + l16) * 40 + half * 8];
            bfr[tn].q[0] = *(const uint4*)bp;          // K = half*8 .. +7
            bfr[tn].q[1] = *(const uint4*)(bp + 16);   // K = 16+half*8 .. +7
        }

        #pragma unroll
        for (int tm = 0; tm < 2; ++tm)
            #pragma unroll
            for (int tn = 0; tn < 4; ++tn)
                acc[tm][tn] = __builtin_amdgcn_wmma_f32_16x16x32_bf16(
                    false, afr[tm].v, false, bfr[tn].v,
                    (short)0, acc[tm][tn], false, false);

        if (more) {
            if constexpr (!ABF) storeAf(Asl[nxt]);
            async_wait0();
            __syncthreads();
        }
    }

    // ---- epilogue (C layout: VGPR i -> M = half*8 + i, lane&15 -> N) ----
    const int rowBase = m0 + waveM * 32 + half * 8;
    const int colBase = n0 + waveN * 64;
    float bv[4];
    #pragma unroll
    for (int tn = 0; tn < 4; ++tn) bv[tn] = bias[colBase + tn * 16 + l16];

#define EPILOOP(GUARDED)                                                       \
    _Pragma("unroll")                                                          \
    for (int tm = 0; tm < 2; ++tm) {                                           \
        const int r0 = rowBase + tm * 16;                                      \
        _Pragma("unroll")                                                      \
        for (int tn = 0; tn < 4; ++tn) {                                       \
            size_t idx = (size_t)r0 * (size_t)N + (colBase + tn * 16 + l16);   \
            _Pragma("unroll")                                                  \
            for (int i = 0; i < 8; ++i) {                                      \
                if (!GUARDED || (r0 + i < M)) {                                \
                    float v = acc[tm][tn][i] + bv[tn];                         \
                    float r;                                                   \
                    if (EPI == 0)      r = fmaxf(v, 0.f);                      \
                    else if (EPI == 1) r = fmaxf((float)((const __bf16*)res)[idx] + v, 0.f); \
                    else               r = (float)((const __bf16*)res)[idx] + fmaxf(v, 0.f); \
                    if (OBF) ((__bf16*)outv)[idx] = (__bf16)r;                 \
                    else     ((float*)outv)[idx]  = r;                         \
                }                                                              \
                idx += (size_t)N;                                              \
            }                                                                  \
        }                                                                      \
    }

    if (m0 + 128 <= M) { EPILOOP(false) } else { EPILOOP(true) }
#undef EPILOOP
}

// ---- weight convert + transpose: src K x N f32 -> dst N x K bf16 ----
__global__ void wconv_kernel(const float* __restrict__ src,
                             unsigned short* __restrict__ dst, int K, int N)
{
    int i = blockIdx.x * 256 + threadIdx.x;
    if (i < K * N) {
        int k = i / N, n = i % N;
        ((__bf16*)dst)[(size_t)n * K + k] = (__bf16)src[i];
    }
}

// ---- flat f32 -> bf16 convert ----
__global__ void cvt_bf16_kernel(const float* __restrict__ src,
                                unsigned short* __restrict__ dst, long long n)
{
    long long i = (long long)blockIdx.x * 256 + threadIdx.x;
    if (i < n) ((__bf16*)dst)[i] = (__bf16)src[i];
}

__global__ void fill0_kernel(float* __restrict__ p, long long n)
{
    long long i = (long long)blockIdx.x * 256 + threadIdx.x;
    if (i < n) p[i] = 0.f;
}

// out[dst[e], :] += val[e] * x[src[e], :]  (XBF: x is bf16, else f32)
template <int XBF>
__global__ void spmm_kernel(const int* __restrict__ src, const int* __restrict__ dst,
                            const float* __restrict__ val,
                            const void* __restrict__ x, float* __restrict__ out)
{
    const int e = blockIdx.x, t = threadIdx.x;
    const int s = src[e], d = dst[e];
    const float v = val[e];
    float xv;
    if (XBF) xv = (float)((const __bf16*)x)[(size_t)s * DIMV + t];
    else     xv = ((const float*)x)[(size_t)s * DIMV + t];
    atomicAdd(&out[(size_t)d * DIMV + t], v * xv);
}

// out[dst[e], :] += x[src[e], :]   (f32)
__global__ void scatter_kernel(const int* __restrict__ src, const int* __restrict__ dst,
                               const float* __restrict__ x, float* __restrict__ out)
{
    const int e = blockIdx.x, t = threadIdx.x;
    const int s = src[e], d = dst[e];
    atomicAdd(&out[(size_t)d * DIMV + t], x[(size_t)s * DIMV + t]);
}

// ---------------------------------------------------------------------------
extern "C" void kernel_launch(void* const* d_in, const int* in_sizes, int n_in,
                              void* d_out, int out_size, void* d_ws, size_t ws_size,
                              hipStream_t stream)
{
    const int NC = in_sizes[0] / DIMV;
    const int NN = in_sizes[1] / DIMV;
    const int E  = in_sizes[4];

    const float* h_cell  = (const float*)d_in[0];
    const float* h_net   = (const float*)d_in[1];
    const float* h1_cell = (const float*)d_in[2];
    const float* h1_net  = (const float*)d_in[3];
    const int*   cn_src  = (const int*)d_in[4];
    const int*   cn_dst  = (const int*)d_in[5];
    const float* cn_val  = (const float*)d_in[6];
    const int*   nc_src  = (const int*)d_in[7];
    const int*   nc_dst  = (const int*)d_in[8];
    const float* nc_val  = (const float*)d_in[9];
    const int*   g_src   = (const int*)d_in[10];
    const int*   g_dst   = (const int*)d_in[11];

    struct WDesc { int idx, K, N; };
    const WDesc wd[14] = {
        {12, 256, 512}, {14, 256, 512},   // 0:r1_W1c  1:r1_W1n
        {16, 512, 256}, {18, 512, 256},   // 2:r1_W2c  3:r1_W2n
        {20, 256, 256}, {22, 256, 256},   // 4:Wn1     5:Wc1
        {24, 256, 256}, {26, 512, 256},   // 6:Wcn1    7:Wcn2
        {28, 256, 256}, {30, 512, 256},   // 8:Wnc1    9:Wnc2
        {32, 256, 512}, {34, 512, 256},   // 10:rn_W1  11:rn_W2
        {36, 256, 512}, {38, 512, 256},   // 12:rc_W1  13:rc_W2
    };

    // ---- workspace bump allocator ----
    char* wp = (char*)d_ws;
    auto alloc = [&](size_t bytes) -> void* {
        void* p = (void*)wp; wp += (bytes + 255) & ~(size_t)255; return p;
    };
    size_t wtot = 0; for (int i = 0; i < 14; ++i) wtot += (size_t)wd[i].K * wd[i].N;
    unsigned short* Wall = (unsigned short*)alloc(wtot * 2);
    unsigned short* Wt[14]; { size_t off = 0;
        for (int i = 0; i < 14; ++i) { Wt[i] = Wall + off; off += (size_t)wd[i].K * wd[i].N; } }

    // bf16 activation buffers
    unsigned short* hc_bf  = (unsigned short*)alloc((size_t)NC * 256 * 2);
    unsigned short* hn_bf  = (unsigned short*)alloc((size_t)NN * 256 * 2);
    unsigned short* h1c_bf = (unsigned short*)alloc((size_t)NC * 256 * 2);
    unsigned short* h1n_bf = (unsigned short*)alloc((size_t)NN * 256 * 2);
    unsigned short* v_c    = (unsigned short*)alloc((size_t)NC * 256 * 2);
    unsigned short* tmp    = (unsigned short*)alloc((size_t)NC * 512 * 2);
    unsigned short* v_nws  = (unsigned short*)alloc((size_t)NN * 256 * 2);
    unsigned short* v_n1   = (unsigned short*)alloc((size_t)NN * 256 * 2);
    unsigned short* v_c1   = (unsigned short*)alloc((size_t)NC * 256 * 2);
    unsigned short* tbuf   = (unsigned short*)alloc((size_t)NC * 256 * 2);
    float*          zbuf   = (float*)alloc((size_t)NC * 256 * 4);   // z then z2

    float* out_c = (float*)d_out;                         // h_N_cell + v_c
    float* out_n = (float*)d_out + (size_t)NC * 256;      // v_n

    // ---- weights: convert + transpose to bf16 N x K ----
    for (int i = 0; i < 14; ++i) {
        int cnt = wd[i].K * wd[i].N;
        wconv_kernel<<<(cnt + 255) / 256, 256, 0, stream>>>(
            (const float*)d_in[wd[i].idx], Wt[i], wd[i].K, wd[i].N);
    }
    // ---- inputs: f32 -> bf16 ----
    auto cvt = [&](const float* s, unsigned short* d, long long n) {
        cvt_bf16_kernel<<<(unsigned)((n + 255) / 256), 256, 0, stream>>>(s, d, n);
    };
    cvt(h_cell,  hc_bf,  (long long)NC * 256);
    cvt(h_net,   hn_bf,  (long long)NN * 256);
    cvt(h1_cell, h1c_bf, (long long)NC * 256);
    cvt(h1_net,  h1n_bf, (long long)NN * 256);

    auto gemm = [&](int epi, int abf, int obf, int M, int N, int K0, int K,
                    const void* A0, const void* A1, const unsigned short* WTp,
                    const float* b, const unsigned short* r, void* o) {
        dim3 g((M + 127) / 128, N / 128), blk(256);
        if      (epi == 0 && abf && obf)  gemm_bf16_wmma<0,1,1><<<g, blk, 0, stream>>>(M, N, K0, K, A0, A1, WTp, b, r, o);
        else if (epi == 0 && !abf && obf) gemm_bf16_wmma<0,0,1><<<g, blk, 0, stream>>>(M, N, K0, K, A0, A1, WTp, b, r, o);
        else if (epi == 1 && abf && obf)  gemm_bf16_wmma<1,1,1><<<g, blk, 0, stream>>>(M, N, K0, K, A0, A1, WTp, b, r, o);
        else if (epi == 1 && abf && !obf) gemm_bf16_wmma<1,1,0><<<g, blk, 0, stream>>>(M, N, K0, K, A0, A1, WTp, b, r, o);
        else if (epi == 2 && abf && obf)  gemm_bf16_wmma<2,1,1><<<g, blk, 0, stream>>>(M, N, K0, K, A0, A1, WTp, b, r, o);
        else                              gemm_bf16_wmma<2,1,0><<<g, blk, 0, stream>>>(M, N, K0, K, A0, A1, WTp, b, r, o);
    };
    auto bias = [&](int slot) { return (const float*)d_in[slot]; };

    // HeteroResidualBlock (cells)
    gemm(0,1,1, NC, 512, 256, 256, hc_bf, nullptr, Wt[0], bias(13), nullptr, tmp);   // hc1
    gemm(1,1,1, NC, 256, 512, 512, tmp,   nullptr, Wt[2], bias(17), hc_bf,  v_c);    // v_c
    // HeteroResidualBlock (nets)
    gemm(0,1,1, NN, 512, 256, 256, hn_bf, nullptr, Wt[1], bias(15), nullptr, tmp);   // hn1
    gemm(1,1,1, NN, 256, 512, 512, tmp,   nullptr, Wt[3], bias(19), hn_bf,  v_nws);  // v_n
    // v_n1, v_c1
    gemm(0,1,1, NN, 256, 256, 256, h1n_bf, nullptr, Wt[4], bias(21), nullptr, v_n1);
    gemm(0,1,1, NC, 256, 256, 256, h1c_bf, nullptr, Wt[5], bias(23), nullptr, v_c1);

    // z = g_cn @ v_c (cells -> nets), f32 accumulate
    {
        long long n = (long long)NN * 256;
        fill0_kernel<<<(unsigned)((n + 255) / 256), 256, 0, stream>>>(zbuf, n);
        spmm_kernel<1><<<E, DIMV, 0, stream>>>(cn_src, cn_dst, cn_val, v_c, zbuf);
    }
    gemm(0,0,1, NN, 256, 256, 256, zbuf, nullptr, Wt[6], bias(25), nullptr, tbuf);   // relu(z@Wcn1+b)
    gemm(2,1,1, NN, 256, 256, 512, v_n1, tbuf,    Wt[7], bias(27), v_nws,  v_nws);   // v_n += relu(concat@Wcn2+b)

    // res_block on v_n -> final v_n (f32) straight into d_out slice
    gemm(0,1,1, NN, 512, 256, 256, v_nws, nullptr, Wt[10], bias(33), nullptr, tmp);  // rn_W1
    gemm(1,1,0, NN, 256, 512, 512, tmp,   nullptr, Wt[11], bias(35), v_nws,  out_n); // rn_W2
    // res_block on v_c (stays bf16)
    gemm(0,1,1, NC, 512, 256, 256, v_c,   nullptr, Wt[12], bias(37), nullptr, tmp);  // rc_W1
    gemm(1,1,1, NC, 256, 512, 512, tmp,   nullptr, Wt[13], bias(39), v_c,    v_c);   // rc_W2

    // z2 = g_nc @ v_n (nets -> cells), v_n is f32 in d_out
    {
        long long n = (long long)NC * 256;
        fill0_kernel<<<(unsigned)((n + 255) / 256), 256, 0, stream>>>(zbuf, n);
        spmm_kernel<0><<<E, DIMV, 0, stream>>>(nc_src, nc_dst, nc_val, out_n, zbuf);
    }
    gemm(0,0,1, NC, 256, 256, 256, zbuf, nullptr, Wt[8], bias(29), nullptr, tbuf);   // relu(z2@Wnc1+b)
    gemm(2,1,0, NC, 256, 256, 512, v_c1, tbuf,    Wt[9], bias(31), v_c,    out_c);   // out_c = v_c + relu(...)

    // h_N_cell: out_c[g_dst] += v_n[g_src]
    scatter_kernel<<<E, DIMV, 0, stream>>>(g_src, g_dst, out_n, out_c);
}
